// Encoder_29205777613055
// MI455X (gfx1250) — matrix operand order repaired and checked by reading, not compiled, not run on hardware
//
#include <hip/hip_runtime.h>

typedef __attribute__((ext_vector_type(16))) __bf16 v16bf;
typedef __attribute__((ext_vector_type(8)))  float  v8f;

#define NDIM 96
#define TILE_M 32          // rows per block (2 row-tiles x 6 col-tiles = 12 waves)
#define GEMM_THREADS 384

// ---------------------------------------------------------------------------
// Degree / normalization
// ---------------------------------------------------------------------------
__global__ __launch_bounds__(256) void init_deg(float* __restrict__ deg, int n) {
    int i = blockIdx.x * blockDim.x + threadIdx.x;
    if (i < n) deg[i] = 1.0f;   // self-loop
}

__global__ __launch_bounds__(256) void count_deg(const int* __restrict__ ei,
                                                 float* __restrict__ deg, int E) {
    int e = blockIdx.x * blockDim.x + threadIdx.x;
    if (e < E) atomicAdd(&deg[ei[E + e]], 1.0f);   // dst row of edge_index
}

__global__ __launch_bounds__(256) void make_dinv(const float* __restrict__ deg,
                                                 float* __restrict__ dinv, int n) {
    int i = blockIdx.x * blockDim.x + threadIdx.x;
    if (i < n) dinv[i] = rsqrtf(deg[i]);           // deg >= 1 always
}

// ---------------------------------------------------------------------------
// WMMA GEMM:  Y[i,:] = ACC[i,:] = dinv[i] * (in[i,:] @ W)
// 32 rows per block, 12 waves; wave (wm,wn) owns the 16x16 tile at
// (row0+16*wm, 16*wn). K=96 in 3 chunks of 32 -> 3 chained v_wmma per wave.
// Safe to run in-place (in == Y): each block stages its 32 rows into LDS
// behind a barrier before any store.
// ---------------------------------------------------------------------------
__global__ __launch_bounds__(GEMM_THREADS) void gcn_gemm_wmma(
    const float* in, const float* __restrict__ Wg,
    const float* __restrict__ dinv, float* Y,
    float* __restrict__ ACC, int nrows)
{
    __shared__ __bf16 Wt[NDIM][NDIM];    // Wt[n][k] = W[k][n]  (K contiguous)
    __shared__ __bf16 Xs[TILE_M][NDIM];  // Xs[m][k]            (K contiguous)

    const int tid  = threadIdx.x;
    const int row0 = blockIdx.x * TILE_M;

    // Stage W transposed into LDS (9216 elems / 384 threads)
    for (int i = tid; i < NDIM * NDIM; i += GEMM_THREADS) {
        int k = i / NDIM, nn = i % NDIM;
        Wt[nn][k] = (__bf16)Wg[i];
    }
    // Stage the 32-row input tile as bf16 (zero-pad past nrows)
    for (int i = tid; i < TILE_M * NDIM; i += GEMM_THREADS) {
        int m = i / NDIM, k = i % NDIM;
        int r = row0 + m;
        Xs[m][k] = (__bf16)((r < nrows) ? in[(size_t)r * NDIM + k] : 0.0f);
    }
    __syncthreads();

    const int wave = tid >> 5;      // 0..11
    const int wm   = wave & 1;      // row tile
    const int wn   = wave >> 1;     // col tile 0..5
    const int lane = tid & 31;
    const int half = lane >> 4;     // 0 | 1
    const int l15  = lane & 15;
    const int m0   = wm * 16;       // row offset inside LDS tile
    const int n0   = wn * 16;       // col offset

    v8f acc = {};
    for (int kc = 0; kc < NDIM; kc += 32) {
        v16bf afrag, bfrag;
        // A 16x32 bf16 layout (ISA 7.12.2): lane l15 = row M; VGPR j holds
        // K pair base = (j<4?0:16) + 8*half + 2*(j&3).
        // B 32x16 layout: lane l15 = col N; VGPR j holds same K pair as A.
#pragma unroll
        for (int j = 0; j < 8; ++j) {
            int ka = kc + ((j < 4) ? 0 : 16) + 8 * half + 2 * (j & 3);
            afrag[2 * j]     = Xs[m0 + l15][ka];
            afrag[2 * j + 1] = Xs[m0 + l15][ka + 1];
            int kb = kc + ((j < 4) ? 0 : 16) + 8 * half + 2 * (j & 3);
            bfrag[2 * j]     = Wt[n0 + l15][kb];
            bfrag[2 * j + 1] = Wt[n0 + l15][kb + 1];
        }
        acc = __builtin_amdgcn_wmma_f32_16x16x32_bf16(
            /*neg_a=*/false, afrag, /*neg_b=*/false, bfrag,
            /*c_mod=*/(short)0, acc, /*reuse_a=*/false, /*reuse_b=*/false);
    }

    // C/D layout: VGPR i -> row = i + 8*half, col = l15 (within tile).
    // Uniform full-tile fast path: straight-line stores, no exec churn.
    const int rbase = row0 + m0 + 8 * half;
    if (row0 + TILE_M <= nrows) {
#pragma unroll
        for (int i = 0; i < 8; ++i) {
            int r = rbase + i;
            float v = acc[i] * dinv[r];
            size_t idx = (size_t)r * NDIM + n0 + l15;
            Y[idx]   = v;   // message values y = dinv * xW
            ACC[idx] = v;   // accumulator seeded with self-loop term
        }
    } else {
#pragma unroll
        for (int i = 0; i < 8; ++i) {
            int r = rbase + i;
            if (r < nrows) {
                float v = acc[i] * dinv[r];
                size_t idx = (size_t)r * NDIM + n0 + l15;
                Y[idx]   = v;
                ACC[idx] = v;
            }
        }
    }
}

// ---------------------------------------------------------------------------
// Edge scatter-add: one wave per edge, 3 coalesced f32 atomics per lane.
// Y is 19.2 MB -> resident in MI455X's 192 MB L2, so gather+atomic stay
// on-chip; HBM only sees the 6.4 MB edge list stream.
// ---------------------------------------------------------------------------
__global__ __launch_bounds__(256) void gcn_edge_agg(
    const int* __restrict__ ei, const float* __restrict__ Y,
    float* __restrict__ ACC, int E)
{
    int gid  = blockIdx.x * blockDim.x + threadIdx.x;
    int e    = gid >> 5;
    int lane = gid & 31;
    if (e < E) {
        int s = ei[e];
        int d = ei[E + e];
        const float* yrow = Y + (size_t)s * NDIM;
        float* arow = ACC + (size_t)d * NDIM;
#pragma unroll
        for (int j = 0; j < 3; ++j) {
            int f = lane + 32 * j;
            atomicAdd(&arow[f], yrow[f]);
        }
    }
}

// ---------------------------------------------------------------------------
// Epilogue: out = prelu(dinv[row] * ACC + b, a), float4-vectorized (96 = 24*4)
// ---------------------------------------------------------------------------
__global__ __launch_bounds__(256) void gcn_epilogue(
    const float4* __restrict__ ACC4, const float* __restrict__ dinv,
    const float* __restrict__ b, const float* __restrict__ a,
    float4* __restrict__ out4, int n)
{
    int i = blockIdx.x * blockDim.x + threadIdx.x;   // over n*24 vec4s
    if (i < n * (NDIM / 4)) {
        int r  = i / (NDIM / 4);
        int c4 = (i % (NDIM / 4)) * 4;
        float di = dinv[r];
        float4 v = ACC4[i];
        float h0 = di * v.x + b[c4 + 0];
        float h1 = di * v.y + b[c4 + 1];
        float h2 = di * v.z + b[c4 + 2];
        float h3 = di * v.w + b[c4 + 3];
        v.x = (h0 > 0.0f) ? h0 : a[c4 + 0] * h0;
        v.y = (h1 > 0.0f) ? h1 : a[c4 + 1] * h1;
        v.z = (h2 > 0.0f) ? h2 : a[c4 + 2] * h2;
        v.w = (h3 > 0.0f) ? h3 : a[c4 + 3] * h3;
        out4[i] = v;
    }
}

// ---------------------------------------------------------------------------
extern "C" void kernel_launch(void* const* d_in, const int* in_sizes, int n_in,
                              void* d_out, int out_size, void* d_ws, size_t ws_size,
                              hipStream_t stream) {
    const float* x  = (const float*)d_in[0];
    const int*   ei = (const int*)  d_in[1];
    const float* W1 = (const float*)d_in[2];
    const float* b1 = (const float*)d_in[3];
    const float* W2 = (const float*)d_in[4];
    const float* b2 = (const float*)d_in[5];
    const float* pa = (const float*)d_in[6];
    float* out = (float*)d_out;

    const int n = in_sizes[0] / NDIM;   // 50000
    const int E = in_sizes[1] / 2;      // 800000

    float* ws   = (float*)d_ws;
    float* deg  = ws;                           // n
    float* dinv = ws + n;                       // n
    float* Y    = ws + 2 * (size_t)n;           // n*96
    float* ACC  = Y + (size_t)n * NDIM;         // n*96

    // Normalization (shared by both layers)
    init_deg <<<(n + 255) / 256, 256, 0, stream>>>(deg, n);
    count_deg<<<(E + 255) / 256, 256, 0, stream>>>(ei, deg, E);
    make_dinv<<<(n + 255) / 256, 256, 0, stream>>>(deg, dinv, n);

    const int gblocks = (n + TILE_M - 1) / TILE_M;          // 1563
    const int eblocks = (E * 32 + 255) / 256;               // wave per edge
    const int pblocks = (n * (NDIM / 4) + 255) / 256;

    // Layer 1
    gcn_gemm_wmma<<<gblocks, GEMM_THREADS, 0, stream>>>(x, W1, dinv, Y, ACC, n);
    gcn_edge_agg <<<eblocks, 256, 0, stream>>>(ei, Y, ACC, E);
    gcn_epilogue <<<pblocks, 256, 0, stream>>>((const float4*)ACC, dinv, b1, pa,
                                               (float4*)Y, n);

    // Layer 2 (GEMM reads Y, writes Y/ACC in place — staged via LDS)
    gcn_gemm_wmma<<<gblocks, GEMM_THREADS, 0, stream>>>(Y, W2, dinv, Y, ACC, n);
    gcn_edge_agg <<<eblocks, 256, 0, stream>>>(ei, Y, ACC, E);
    gcn_epilogue <<<pblocks, 256, 0, stream>>>((const float4*)ACC, dinv, b2, pa,
                                               (float4*)out, n);
}
